// BridgeAttention_55104430408031
// MI455X (gfx1250) — matrix operand order repaired and checked by reading, not compiled, not run on hardware
//
#include <hip/hip_runtime.h>
#include <hip/hip_bf16.h>

// ---------------------------------------------------------------------------
// CDNA5 (gfx1250) bf16 WMMA attention + bridge detector.
// B=2, T=2048, C=1024, H=16, D=64. wave32, 256-thread (8-wave) workgroups.
// Round 2: K-tile staging now uses GLOBAL_LOAD_ASYNC_TO_LDS_B128 (ASYNCcnt).
// ---------------------------------------------------------------------------

typedef __bf16 v16bf __attribute__((ext_vector_type(16)));
typedef float  v8f   __attribute__((ext_vector_type(8)));

union AFrag {               // one 16x32 (A) or 32x16 (B) bf16 fragment
    v16bf v;
    unsigned int u[8];
    unsigned short h[16];
};

__device__ __forceinline__ unsigned short f2bf(float f) {
    unsigned int u = __float_as_uint(f);
    unsigned int r = (u + 0x7FFFu + ((u >> 16) & 1u)) >> 16;  // RNE
    return (unsigned short)r;
}

__device__ __forceinline__ v8f v8f_zero() {
    v8f z;
#pragma unroll
    for (int i = 0; i < 8; ++i) z[i] = 0.0f;
    return z;
}

__device__ __forceinline__ v8f wmma_bf16(const AFrag& a, const AFrag& b, v8f c) {
    return __builtin_amdgcn_wmma_f32_16x16x32_bf16(
        false, a.v, false, b.v, (short)0, c, false, false);
}

// generic pointer -> 32-bit LDS byte offset (addrspace(3) value)
__device__ __forceinline__ unsigned int lds_offset(const void* p) {
    return (unsigned int)(unsigned long long)
        (__attribute__((address_space(3))) const void*)p;
}

// async copy of 16 bytes global -> LDS, tracked on ASYNCcnt
__device__ __forceinline__ void async_copy_b128(unsigned int lds_off,
                                                const void* gptr) {
    asm volatile("global_load_async_to_lds_b128 %0, %1, off"
                 :: "v"(lds_off), "v"((unsigned long long)gptr)
                 : "memory");
}

__device__ __forceinline__ void wait_asynccnt0() {
    asm volatile("s_wait_asynccnt 0x0" ::: "memory");
}

// Problem constants
#define TT 2048
#define CC 1024
#define HH 16
#define DD 64
#define QKVH 4194304      // B*H*T*D halfs per tensor

// ---------------------------------------------------------------------------
// Kernel 1: q/k/v projections.  x[4096,1024] @ W[1024,1024] + b -> bf16
// scattered into [B,H,T,D].  q is pre-scaled by 1/sqrt(D).
// Tile: 128x128 per workgroup, per-wave 64x32 (4x2 WMMA tiles), K step 32.
// ---------------------------------------------------------------------------
__global__ __launch_bounds__(256)
void qkv_proj_kernel(const float* __restrict__ x,
                     const float* __restrict__ Wq, const float* __restrict__ bq,
                     const float* __restrict__ Wk, const float* __restrict__ bk,
                     const float* __restrict__ Wv, const float* __restrict__ bv,
                     unsigned short* __restrict__ qo,
                     unsigned short* __restrict__ ko,
                     unsigned short* __restrict__ vo) {
    __shared__ unsigned short lA[128][34];   // [m][k]  k-contiguous
    __shared__ unsigned short lB[128][34];   // [n][k]  k-contiguous (W transposed)

    const int which = blockIdx.z;
    const float* W;  const float* bias;  unsigned short* out;  float scale;
    if (which == 0)      { W = Wq; bias = bq; out = qo; scale = 0.125f; }  // 1/sqrt(64)
    else if (which == 1) { W = Wk; bias = bk; out = ko; scale = 1.0f;   }
    else                 { W = Wv; bias = bv; out = vo; scale = 1.0f;   }

    const int M0 = blockIdx.y * 128;
    const int N0 = blockIdx.x * 128;
    const int tid  = threadIdx.x;
    const int lane = tid & 31;
    const int wid  = tid >> 5;
    const int wm = (wid >> 2) * 64;          // 2 wave rows of 64
    const int wn = (wid & 3) * 32;           // 4 wave cols of 32
    const int l15   = lane & 15;
    const int koffA = (lane < 16) ? 0 : 8;   // A-frag K offset (halfs)
    const int koffB = (lane < 16) ? 0 : 16;  // B-frag K offset (halfs)
    const int rbase = (lane < 16) ? 0 : 8;   // C/D row offset

    v8f acc[4][2];
#pragma unroll
    for (int mi = 0; mi < 4; ++mi)
#pragma unroll
        for (int ni = 0; ni < 2; ++ni) acc[mi][ni] = v8f_zero();

    for (int kk = 0; kk < CC; kk += 32) {
        __syncthreads();
        {   // stage A: 128 rows x 32 k, fp32 -> bf16
            const int r  = tid >> 1;
            const int c0 = (tid & 1) * 16;
            const float* src = x + (M0 + r) * CC + kk + c0;
#pragma unroll
            for (int i = 0; i < 16; ++i) lA[r][c0 + i] = f2bf(src[i]);
        }
        {   // stage B transposed: lB[n][k] = W[kk+k][N0+n]
            const int kr  = tid >> 3;
            const int n0l = (tid & 7) * 16;
            const float* src = W + (kk + kr) * CC + N0 + n0l;
#pragma unroll
            for (int i = 0; i < 16; ++i) lB[n0l + i][kr] = f2bf(src[i]);
        }
        __syncthreads();

        AFrag a[4], bfr[2];
#pragma unroll
        for (int mi = 0; mi < 4; ++mi) {
            const int row = wm + 16 * mi + l15;
#pragma unroll
            for (int j = 0; j < 8; ++j) {
                const int kh = ((j < 4) ? 2 * j : 2 * j + 8) + koffA;
                a[mi].u[j] = *(const unsigned int*)&lA[row][kh];
            }
        }
#pragma unroll
        for (int ni = 0; ni < 2; ++ni) {
            const int col = wn + 16 * ni + l15;
#pragma unroll
            for (int j = 0; j < 8; ++j)
                bfr[ni].u[j] = *(const unsigned int*)&lB[col][((j < 4) ? 2 * j : 2 * j + 8) + koffA];
        }
#pragma unroll
        for (int mi = 0; mi < 4; ++mi)
#pragma unroll
            for (int ni = 0; ni < 2; ++ni)
                acc[mi][ni] = wmma_bf16(a[mi], bfr[ni], acc[mi][ni]);
    }

    // epilogue: bias, scale, scatter to [B,H,T,D] bf16
#pragma unroll
    for (int mi = 0; mi < 4; ++mi) {
#pragma unroll
        for (int ni = 0; ni < 2; ++ni) {
            const int n_g = N0 + wn + 16 * ni + l15;
            const float bb = bias[n_g];
            const int hh = n_g >> 6;
            const int dd = n_g & 63;
#pragma unroll
            for (int r = 0; r < 8; ++r) {
                const int m_g = M0 + wm + 16 * mi + r + rbase;
                const int bi = m_g >> 11;          // / T
                const int t  = m_g & (TT - 1);
                const float val = (acc[mi][ni][r] + bb) * scale;
                out[(((bi * HH + hh) * TT) + t) * DD + dd] = f2bf(val);
            }
        }
    }
}

// ---------------------------------------------------------------------------
// Kernel 2: flash attention.  One workgroup per (b,h, 128-query block);
// each of the 8 waves owns a 16-query tile with online softmax.
// K super-block staged via GLOBAL_LOAD_ASYNC_TO_LDS_B128 (ASYNCcnt);
// V staged transposed through VGPRs (async path cannot transpose).
// ---------------------------------------------------------------------------
__global__ __launch_bounds__(256)
void flash_attn_kernel(const unsigned short* __restrict__ q,
                       const unsigned short* __restrict__ kmat,
                       const unsigned short* __restrict__ vmat,
                       unsigned short* __restrict__ attn_out) {
    // 72-half row stride (144 B): 16B-aligned rows for B128 async stores
    __shared__ __attribute__((aligned(16))) unsigned short sK[128][72];
    __shared__ unsigned short sVt[64][130];      // [d][key]   key-contiguous
    __shared__ unsigned short sS[8][16][34];     // per-wave P relayout scratch

    const int bh  = blockIdx.y;                  // b*H + h
    const int b   = bh >> 4;
    const int h   = bh & 15;
    const int blk = blockIdx.x;                  // 128-query block
    const int tid  = threadIdx.x;
    const int lane = tid & 31;
    const int wid  = tid >> 5;
    const int l15   = lane & 15;
    const int rbase = (lane < 16) ? 0 : 8;
    const int koffA = (lane < 16) ? 0 : 8;
    const int koffB = (lane < 16) ? 0 : 16;
    const int tq = blk * 128 + wid * 16;         // this wave's first query row

    const unsigned short* Qb = q    + (size_t)bh * TT * DD;
    const unsigned short* Kb = kmat + (size_t)bh * TT * DD;
    const unsigned short* Vb = vmat + (size_t)bh * TT * DD;

    // Q fragments (16x64 = two 16x32 A-frags), q already scaled by 1/sqrt(D)
    AFrag qa[2];
#pragma unroll
    for (int f = 0; f < 2; ++f) {
        const unsigned short* src = Qb + (tq + l15) * DD + 32 * f;
#pragma unroll
        for (int j = 0; j < 8; ++j) {
            const int kh = ((j < 4) ? 2 * j : 2 * j + 8) + koffA;
            qa[f].u[j] = *(const unsigned int*)&src[kh];
        }
    }

    v8f o[4];
#pragma unroll
    for (int ni = 0; ni < 4; ++ni) o[ni] = v8f_zero();
    float mrow[8], lrow[8];
#pragma unroll
    for (int r = 0; r < 8; ++r) { mrow[r] = -3.0e38f; lrow[r] = 0.0f; }

    for (int sb = 0; sb <= blk; ++sb) {
        const int ksb = sb * 128;
        __syncthreads();
        {   // ---- K tile: async global -> LDS, 4 x b128 per thread
            // transfer index i in [0,1024): row = i>>3, 16B segment = i&7
#pragma unroll
            for (int t4 = 0; t4 < 4; ++t4) {
                const int i   = tid * 4 + t4;
                const int row = i >> 3;
                const int seg = i & 7;
                async_copy_b128(lds_offset(&sK[row][seg * 8]),
                                Kb + (ksb + row) * DD + seg * 8);
            }
            // ---- V tile: manual transpose into sVt[d][key]
            const int r  = tid >> 1;
            const int c0 = (tid & 1) * 32;
            const unsigned short* vs = Vb + (ksb + r) * DD + c0;
#pragma unroll
            for (int i = 0; i < 32; ++i) sVt[c0 + i][r] = vs[i];
            if (sb < blk) {   // prefetch next super-block
                __builtin_prefetch(Kb + (ksb + 128 + r) * DD + c0, 0, 3);
                __builtin_prefetch(Vb + (ksb + 128 + r) * DD + c0, 0, 3);
            }
        }
        wait_asynccnt0();        // our async K transfers landed in LDS
        __syncthreads();         // everyone's transfers visible to all waves

        for (int j4 = 0; j4 < 4; ++j4) {
            const int kb = ksb + 32 * j4;        // key block of 32
            if (kb > tq + 15) break;             // causal: nothing visible
            const int kl = 32 * j4;              // local key base in LDS tile

            // ---- S = Q @ K^T : two 16x16 tiles (keys kb..kb+15, kb+16..kb+31)
            v8f s0 = v8f_zero(), s1 = v8f_zero();
#pragma unroll
            for (int f = 0; f < 2; ++f) {
                AFrag bk0, bk1;
                const unsigned short* s0p = &sK[kl + l15][32 * f];
                const unsigned short* s1p = &sK[kl + 16 + l15][32 * f];
#pragma unroll
                for (int j = 0; j < 8; ++j) {
                    bk0.u[j] = *(const unsigned int*)&s0p[((j < 4) ? 2 * j : 2 * j + 8) + koffA];
                    bk1.u[j] = *(const unsigned int*)&s1p[((j < 4) ? 2 * j : 2 * j + 8) + koffA];
                }
                s0 = wmma_bf16(qa[f], bk0, s0);
                s1 = wmma_bf16(qa[f], bk1, s1);
            }

            // ---- causal mask (diagonal blocks only)
            if (kb + 31 > tq) {
#pragma unroll
                for (int r = 0; r < 8; ++r) {
                    const int qrow = tq + r + rbase;
                    const int key0 = kb + l15;
                    if (key0 > qrow)      s0[r] = -1.0e30f;
                    if (key0 + 16 > qrow) s1[r] = -1.0e30f;
                }
            }

            // ---- online softmax row stats (reduce across 16-lane halves)
            float pmax[8];
#pragma unroll
            for (int r = 0; r < 8; ++r) pmax[r] = fmaxf(s0[r], s1[r]);
#pragma unroll
            for (int m = 1; m < 16; m <<= 1)
#pragma unroll
                for (int r = 0; r < 8; ++r)
                    pmax[r] = fmaxf(pmax[r], __shfl_xor(pmax[r], m, 32));

            float al[8];
#pragma unroll
            for (int r = 0; r < 8; ++r) {
                const float mn = fmaxf(mrow[r], pmax[r]);
                al[r] = __expf(mrow[r] - mn);
                mrow[r] = mn;
            }
            float psum[8];
#pragma unroll
            for (int r = 0; r < 8; ++r) {
                s0[r] = __expf(s0[r] - mrow[r]);
                s1[r] = __expf(s1[r] - mrow[r]);
                psum[r] = s0[r] + s1[r];
            }
#pragma unroll
            for (int m = 1; m < 16; m <<= 1)
#pragma unroll
                for (int r = 0; r < 8; ++r)
                    psum[r] += __shfl_xor(psum[r], m, 32);
#pragma unroll
            for (int r = 0; r < 8; ++r) lrow[r] = lrow[r] * al[r] + psum[r];
#pragma unroll
            for (int ni = 0; ni < 4; ++ni)
#pragma unroll
                for (int r = 0; r < 8; ++r) o[ni][r] *= al[r];

            // ---- relayout P (C-frag) -> A-frag via per-wave LDS scratch
            unsigned short (&ss)[16][34] = sS[wid];
#pragma unroll
            for (int r = 0; r < 8; ++r) {
                ss[r + rbase][l15]      = f2bf(s0[r]);
                ss[r + rbase][16 + l15] = f2bf(s1[r]);
            }
            asm volatile("" ::: "memory");       // keep DS store->load order
            AFrag pa;
#pragma unroll
            for (int j = 0; j < 8; ++j) {
                const int kh = ((j < 4) ? 2 * j : 2 * j + 8) + koffA;
                pa.u[j] = *(const unsigned int*)&ss[l15][kh];
            }

            // ---- O += P(16x32) @ V(32x64) : 4 d-tiles
#pragma unroll
            for (int ni = 0; ni < 4; ++ni) {
                AFrag bv_;
                const unsigned short* src = &sVt[ni * 16 + l15][kl];
#pragma unroll
                for (int j = 0; j < 8; ++j)
                    bv_.u[j] = *(const unsigned int*)&src[((j < 4) ? 2 * j : 2 * j + 8) + koffA];
                o[ni] = wmma_bf16(pa, bv_, o[ni]);
            }
        }
    }

    // ---- normalize and store attn output as bf16 [B,T,C]
    float inv[8];
#pragma unroll
    for (int r = 0; r < 8; ++r) inv[r] = 1.0f / lrow[r];
#pragma unroll
    for (int ni = 0; ni < 4; ++ni) {
        const int c = h * DD + ni * 16 + l15;
#pragma unroll
        for (int r = 0; r < 8; ++r) {
            const int t = tq + r + rbase;
            attn_out[((size_t)b * TT + t) * CC + c] = f2bf(o[ni][r] * inv[r]);
        }
    }
}

// ---------------------------------------------------------------------------
// Kernel 3: output projection. attn_bf16[4096,1024] @ Wp + bp -> y fp32.
// ---------------------------------------------------------------------------
__global__ __launch_bounds__(256)
void out_proj_kernel(const unsigned short* __restrict__ attn,
                     const float* __restrict__ Wp, const float* __restrict__ bp,
                     float* __restrict__ y) {
    __shared__ unsigned short lA[128][34];
    __shared__ unsigned short lB[128][34];

    const int M0 = blockIdx.y * 128;
    const int N0 = blockIdx.x * 128;
    const int tid  = threadIdx.x;
    const int lane = tid & 31;
    const int wid  = tid >> 5;
    const int wm = (wid >> 2) * 64;
    const int wn = (wid & 3) * 32;
    const int l15   = lane & 15;
    const int koffA = (lane < 16) ? 0 : 8;
    const int koffB = (lane < 16) ? 0 : 16;
    const int rbase = (lane < 16) ? 0 : 8;

    v8f acc[4][2];
#pragma unroll
    for (int mi = 0; mi < 4; ++mi)
#pragma unroll
        for (int ni = 0; ni < 2; ++ni) acc[mi][ni] = v8f_zero();

    for (int kk = 0; kk < CC; kk += 32) {
        __syncthreads();
        {   // A already bf16: straight dword copies
            const int r  = tid >> 1;
            const int c0 = (tid & 1) * 16;
            const unsigned short* src = attn + (M0 + r) * CC + kk + c0;
#pragma unroll
            for (int i = 0; i < 16; i += 2)
                *(unsigned int*)&lA[r][c0 + i] = *(const unsigned int*)&src[i];
        }
        {
            const int kr  = tid >> 3;
            const int n0l = (tid & 7) * 16;
            const float* src = Wp + (kk + kr) * CC + N0 + n0l;
#pragma unroll
            for (int i = 0; i < 16; ++i) lB[n0l + i][kr] = f2bf(src[i]);
        }
        __syncthreads();

        AFrag a[4], bfr[2];
#pragma unroll
        for (int mi = 0; mi < 4; ++mi) {
            const int row = wm + 16 * mi + l15;
#pragma unroll
            for (int j = 0; j < 8; ++j) {
                const int kh = ((j < 4) ? 2 * j : 2 * j + 8) + koffA;
                a[mi].u[j] = *(const unsigned int*)&lA[row][kh];
            }
        }
#pragma unroll
        for (int ni = 0; ni < 2; ++ni) {
            const int col = wn + 16 * ni + l15;
#pragma unroll
            for (int j = 0; j < 8; ++j)
                bfr[ni].u[j] = *(const unsigned int*)&lB[col][((j < 4) ? 2 * j : 2 * j + 8) + koffA];
        }
#pragma unroll
        for (int mi = 0; mi < 4; ++mi)
#pragma unroll
            for (int ni = 0; ni < 2; ++ni)
                acc[mi][ni] = wmma_bf16(a[mi], bfr[ni], acc[mi][ni]);
    }

#pragma unroll
    for (int mi = 0; mi < 4; ++mi) {
#pragma unroll
        for (int ni = 0; ni < 2; ++ni) {
            const int n_g = N0 + wn + 16 * ni + l15;
            const float bb = bp[n_g];
#pragma unroll
            for (int r = 0; r < 8; ++r) {
                const int m_g = M0 + wm + 16 * mi + r + rbase;
                y[(size_t)m_g * CC + n_g] = acc[mi][ni][r] + bb;
            }
        }
    }
}

// ---------------------------------------------------------------------------
// Kernel 4: bridge detector on last token's selected neurons.
// ---------------------------------------------------------------------------
__global__ __launch_bounds__(128)
void bridge_kernel(const float* __restrict__ y,
                   const float* __restrict__ Wd, const float* __restrict__ bd,
                   const int* __restrict__ idx,
                   float* __restrict__ out_flag, float* __restrict__ out_act) {
    __shared__ float partial[128];
    const int tid = threadIdx.x;
    for (int b = 0; b < 2; ++b) {
        float p = 0.0f;
        if (tid < 102) {
            const float a = y[((size_t)b * TT + (TT - 1)) * CC + idx[tid]];
            out_act[b * 102 + tid] = a;
            p = a * Wd[tid];
        }
        partial[tid] = p;
        __syncthreads();
        if (tid < 64) partial[tid] += partial[tid + 64];
        __syncthreads();
        if (tid < 32) {
            float s = partial[tid] + partial[tid + 32];
#pragma unroll
            for (int m = 16; m >= 1; m >>= 1) s += __shfl_xor(s, m, 32);
            if (tid == 0) {
                const float sig = 1.0f / (1.0f + __expf(-(s + bd[0])));
                out_flag[b] = (sig > 0.5f) ? 1.0f : 0.0f;
            }
        }
        __syncthreads();
    }
}

// ---------------------------------------------------------------------------
extern "C" void kernel_launch(void* const* d_in, const int* in_sizes, int n_in,
                              void* d_out, int out_size, void* d_ws, size_t ws_size,
                              hipStream_t stream) {
    const float* x  = (const float*)d_in[0];
    const float* Wq = (const float*)d_in[1];
    const float* bq = (const float*)d_in[2];
    const float* Wk = (const float*)d_in[3];
    const float* bk = (const float*)d_in[4];
    const float* Wv = (const float*)d_in[5];
    const float* bv = (const float*)d_in[6];
    const float* Wp = (const float*)d_in[7];
    const float* bp = (const float*)d_in[8];
    const float* Wd = (const float*)d_in[9];
    const float* bd = (const float*)d_in[10];
    const int* bridge_idx = (const int*)d_in[11];

    // workspace: q,k,v,attn as bf16 (8 MB each = 32 MB total)
    unsigned short* qb = (unsigned short*)d_ws;
    unsigned short* kb = qb + QKVH;
    unsigned short* vb = kb + QKVH;
    unsigned short* ab = vb + QKVH;

    float* y        = (float*)d_out;            // [B,T,C] = 4,194,304 floats
    float* out_flag = y + (size_t)2 * TT * CC;  // [B,1] as 0/1 floats
    float* out_act  = out_flag + 2;             // [B,102]

    qkv_proj_kernel<<<dim3(8, 32, 3), 256, 0, stream>>>(
        x, Wq, bq, Wk, bk, Wv, bv, qb, kb, vb);
    flash_attn_kernel<<<dim3(16, 32), 256, 0, stream>>>(qb, kb, vb, ab);
    out_proj_kernel<<<dim3(8, 32), 256, 0, stream>>>(ab, Wp, bp, y);
    bridge_kernel<<<1, 128, 0, stream>>>(y, Wd, bd, bridge_idx, out_flag, out_act);
}
